// StochasticTwoLayerGCN_21981642620983
// MI455X (gfx1250) — matrix operand order repaired and checked by reading, not compiled, not run on hardware
//
#include <hip/hip_runtime.h>
#include <hip/hip_bf16.h>
#include <math.h>

typedef _Float16 v16h __attribute__((ext_vector_type(16)));
typedef float    v8f  __attribute__((ext_vector_type(8)));

// Guaranteed native CDNA5 f32 atomic add (no-return form, tracked by STOREcnt).
// Avoids any chance of clang expanding fadd into a CAS loop.
__device__ __forceinline__ void atomicAddF(float* p, float v) {
    asm volatile("global_atomic_add_f32 %0, %1, off"
                 :
                 : "v"(p), "v"(v)
                 : "memory");
}

// ---------------------------------------------------------------- utilities
__global__ void zero_kernel(float* __restrict__ p, int n) {
    int i = blockIdx.x * blockDim.x + threadIdx.x;
    if (i < n) p[i] = 0.0f;
}

// Count degrees: outdeg[src[e]] += 1, indeg[dst[e]] += 1
__global__ void degree_kernel(const int* __restrict__ src, const int* __restrict__ dst,
                              float* __restrict__ outdeg, float* __restrict__ indeg, int E) {
    int e = blockIdx.x * blockDim.x + threadIdx.x;
    if (e >= E) return;
    atomicAddF(outdeg + src[e], 1.0f);
    atomicAddF(indeg  + dst[e], 1.0f);
}

// In-place deg -> clip(deg,1)^-0.5
__global__ void rsqrt_kernel(float* __restrict__ p, int n) {
    int i = blockIdx.x * blockDim.x + threadIdx.x;
    if (i < n) p[i] = rsqrtf(fmaxf(p[i], 1.0f));
}

// Scatter-add SpMM with source-side norm folded in:
//   agg[dst[e], f] += x[src[e], f] * rout[src[e]]
// F/4 lanes per edge (SHIFT = log2(F/4)); each lane moves a float4.
template <int F, int SHIFT>
__global__ void scatter_kernel(const float* __restrict__ xin,
                               const int* __restrict__ src, const int* __restrict__ dst,
                               const float* __restrict__ rout, float* __restrict__ agg,
                               int E) {
    unsigned gid = blockIdx.x * blockDim.x + threadIdx.x;
    int e = gid >> SHIFT;
    if (e >= E) return;
    int f = (gid & ((1u << SHIFT) - 1u)) << 2;
    int s = src[e];
    int d = dst[e];
    float r = rout[s];
    const float4 v = *(const float4*)(xin + (size_t)s * F + f);
    float* ap = agg + (size_t)d * F + f;
    atomicAddF(ap + 0, v.x * r);
    atomicAddF(ap + 1, v.y * r);
    atomicAddF(ap + 2, v.z * r);
    atomicAddF(ap + 3, v.w * r);
}

// Dense GEMM with WMMA: out = relu((A * rA[row]) @ W + bias)
// A: [M,K] f32 (row-scaled by rA = in_deg^-0.5), W: [K,N] f32, out: [M,N] f32.
// One 16x16 output tile per wave32; f16 fragments, f32 accumulation.
// K, N compile-time -> K-loop fully unrolled (K/32 v_wmma per tile).
// M % 16 == 0 holds for all shapes here.
template <int K, int N>
__global__ void gemm_wmma_relu_kernel(const float* __restrict__ A,
                                      const float* __restrict__ rA,
                                      const float* __restrict__ W,
                                      const float* __restrict__ bias,
                                      float* __restrict__ out,
                                      int M) {
    constexpr int tilesN = N / 16;
    const int lane    = threadIdx.x & 31;
    const int wave    = threadIdx.x >> 5;
    const int nTiles  = (M >> 4) * tilesN;
    const int tileIdx = blockIdx.x * (blockDim.x >> 5) + wave;
    if (tileIdx >= nTiles) return;   // uniform per wave: EXEC stays all-ones for WMMA

    const int tm  = (tileIdx / tilesN) << 4;
    const int tn  = (tileIdx % tilesN) << 4;
    const int hi  = lane >> 4;       // lane group 0/1
    const int l15 = lane & 15;

    // A fragment row for this lane (16-bit A 16x32 layout: lane%16 = M row,
    // lane group selects K sub-chunks {0..7,16..23} vs {8..15,24..31})
    const float* arow  = A + (size_t)(tm + l15) * K;
    const float  scale = rA[tm + l15];
    // B fragment column (B 32x16: N = lane%16, lane group selects K sub-chunks {0..7,16..23} vs {8..15,24..31})
    const float* wcol = W + tn + l15;

    v8f c = {};
#pragma unroll
    for (int k0 = 0; k0 < K; k0 += 32) {
        v16h a, b;
        const int ka = k0 + hi * 8;
#pragma unroll
        for (int i = 0; i < 8; ++i) {
            a[i]     = (_Float16)(arow[ka + i]      * scale);
            a[8 + i] = (_Float16)(arow[ka + 16 + i] * scale);
        }
        const int kb = k0 + hi * 8;
#pragma unroll
        for (int j = 0; j < 16; ++j) {
            b[j] = (_Float16)wcol[(kb + j + (j >= 8 ? 8 : 0)) * N];
        }
        // 8 args: (neg_a, A, neg_b, B, c_mod, C, reuse_a, reuse_b)
        c = __builtin_amdgcn_wmma_f32_16x16x32_f16(false, a, false, b, (short)0, c,
                                                   false, false);
    }

    // C/D 16x16 f32 layout: VGPR v -> row v (+8 for lane group 1), col = lane%16
    const float bv = bias[tn + l15];
    float* obase = out + (size_t)(tm + hi * 8) * N + tn + l15;
#pragma unroll
    for (int v = 0; v < 8; ++v) {
        float val = c[v] + bv;
        val = val > 0.0f ? val : 0.0f;
        obase[(size_t)v * N] = val;
    }
}

extern "C" void kernel_launch(void* const* d_in, const int* in_sizes, int n_in,
                              void* d_out, int out_size, void* d_ws, size_t ws_size,
                              hipStream_t stream) {
    (void)n_in; (void)out_size; (void)ws_size;
    const float* x   = (const float*)d_in[0];
    const float* W1  = (const float*)d_in[1];
    const float* b1  = (const float*)d_in[2];
    const float* W2  = (const float*)d_in[3];
    const float* b2  = (const float*)d_in[4];
    const int* src1  = (const int*)d_in[5];
    const int* dst1  = (const int*)d_in[6];
    const int* src2  = (const int*)d_in[7];
    const int* dst2  = (const int*)d_in[8];

    const int F_in = 128, F_hid = 256, F_out = 128;
    const int N0 = in_sizes[0] / F_in;   // 200000
    const int N1 = 100000;
    const int N2 = 50000;
    const int E1 = in_sizes[5];          // 1600000
    const int E2 = in_sizes[7];          // 800000

    // Workspace layout (f32)
    float* ws      = (float*)d_ws;
    float* outdeg1 = ws;                                 // N0, becomes rout1
    float* indeg1  = ws + N0;                            // N1, becomes rin1
    float* outdeg2 = ws + N0 + N1;                       // N1, becomes rout2
    float* indeg2  = ws + N0 + 2 * N1;                   // N2, becomes rin2
    const int degTot = N0 + 2 * N1 + N2;                 // 450000
    float* agg     = ws + degTot;                        // max(N1*128, N2*256) = 12.8M, reused
    const int aggSz = N1 * F_in;                         // == N2 * F_hid
    float* h       = agg + aggSz;                        // N1*256 = 25.6M

    const int T = 256;

    // 1) zero degree counters + agg
    zero_kernel<<<(degTot + aggSz + T - 1) / T, T, 0, stream>>>(ws, degTot + aggSz);
    // 2) degrees
    degree_kernel<<<(E1 + T - 1) / T, T, 0, stream>>>(src1, dst1, outdeg1, indeg1, E1);
    degree_kernel<<<(E2 + T - 1) / T, T, 0, stream>>>(src2, dst2, outdeg2, indeg2, E2);
    // 3) deg -> clip(deg,1)^-0.5 in place
    rsqrt_kernel<<<(degTot + T - 1) / T, T, 0, stream>>>(ws, degTot);
    // 4) layer-1 scatter: agg += x[src] * rout1[src]   (32 lanes/edge, F=128)
    {
        unsigned total = (unsigned)E1 * 32u;
        scatter_kernel<128, 5><<<(total + T - 1) / T, T, 0, stream>>>(
            x, src1, dst1, outdeg1, agg, E1);
    }
    // 5) h = relu((agg * rin1) @ W1 + b1)  [100000 x 256]
    {
        int nTiles = (N1 / 16) * (F_hid / 16);           // 100000
        gemm_wmma_relu_kernel<128, 256><<<(nTiles + 7) / 8, T, 0, stream>>>(
            agg, indeg1, W1, b1, h, N1);
    }
    // 6) re-zero agg for layer 2
    zero_kernel<<<(aggSz + T - 1) / T, T, 0, stream>>>(agg, aggSz);
    // 7) layer-2 scatter: agg += h[src] * rout2[src]   (64 lanes/edge, F=256)
    {
        unsigned total = (unsigned)E2 * 64u;
        scatter_kernel<256, 6><<<(total + T - 1) / T, T, 0, stream>>>(
            h, src2, dst2, outdeg2, agg, E2);
    }
    // 8) out = relu((agg * rin2) @ W2 + b2)  [50000 x 128]
    {
        int nTiles = (N2 / 16) * (F_out / 16);           // 25000
        gemm_wmma_relu_kernel<256, 128><<<(nTiles + 7) / 8, T, 0, stream>>>(
            agg, indeg2, W2, b2, (float*)d_out, N2);
    }
}
